// LinearTransformer_17970143167201
// MI455X (gfx1250) — compile-verified, matrix operand order checked by reading, not run on hardware
//
#include <hip/hip_runtime.h>
#include <math.h>

// ---------------------------------------------------------------------------
// MI455X (gfx1250) linear-attention transformer encoder.
// All heavy math via v_wmma_f32_16x16x32_bf16; every WMMA fragment is built
// from two contiguous 16-byte LDS reads (ds_load_b128). bf16 conversions use
// the native gfx1250 convert instructions (plain __bf16 casts).
// ---------------------------------------------------------------------------

typedef __bf16 bf16;
typedef __attribute__((ext_vector_type(16))) __bf16 v16bf;
typedef __attribute__((ext_vector_type(8)))  __bf16 v8bf;
typedef __attribute__((ext_vector_type(8)))  float  v8f;

#define LN_EPS 1e-5f
#define ATT_EPS 1e-6f

__device__ __forceinline__ bf16 f2bf(float f) { return (bf16)f; }   // native cvt
__device__ __forceinline__ float bf2f(bf16 b) { return (float)b; }  // native cvt

__device__ __forceinline__ v8f v8f_zero() {
  v8f z;
#pragma unroll
  for (int i = 0; i < 8; ++i) z[i] = 0.f;
  return z;
}
// 16-bit fragment layout (CDNA5 ISA): per lane, elements 0..7 are
// contraction indices khalf+0..7, elements 8..15 are khalf+16..23. With the
// contraction dim innermost in LDS this is two contiguous b128 loads.
__device__ __forceinline__ v16bf frag_ld(const bf16* p) {
  union { v16bf v; v8bf h[2]; } u;
  u.h[0] = *(const v8bf*)(p);
  u.h[1] = *(const v8bf*)(p + 16);
  return u.v;
}

// ---------------------------------------------------------------------------
// GEMM: C[M,N] = act(A[M,K] @ W[K,N] + bias) (+ res). Row-major global.
// 256 threads (8 waves), block tile 128x128, staged K-panel 64 (2 wmma steps
// per barrier pair). Wave (wm,wn) owns 32x64 = 2x4 wmma accumulators.
// ACT: 0=none, 1=elu+1 (phi feature map), 2=exact gelu.  RES: add residual.
// ---------------------------------------------------------------------------
#define BM 128
#define BN 128
#define BK 64
#define GPITCH (BK + 8)   // 72 bf16 = 144B rows (16B aligned)

template <int ACT, bool RES>
__global__ __launch_bounds__(256)
void gemm_bias_act_wmma(const float* __restrict__ A, const float* __restrict__ W,
                        const float* __restrict__ bias, const float* __restrict__ res,
                        float* __restrict__ C, int M, int N, int K) {
  __shared__ bf16 As[BM][GPITCH];    // A tile, k innermost
  __shared__ bf16 BsT[BN][GPITCH];   // B tile TRANSPOSED: BsT[n][k]

  const int tid   = threadIdx.x;
  const int lane  = tid & 31;
  const int wid   = tid >> 5;
  const int wm    = wid >> 1;           // 0..3 : 32-row strip
  const int wn    = wid & 1;            // 0..1 : 64-col strip
  const int lrow  = lane & 15;
  const int khalf = (lane >> 4) * 8;
  const int rowBase = blockIdx.y * BM;
  const int colBase = blockIdx.x * BN;

  v8f acc[2][4];
#pragma unroll
  for (int i = 0; i < 2; ++i)
#pragma unroll
    for (int j = 0; j < 4; ++j) acc[i][j] = v8f_zero();

  const int arow = tid >> 1, acol = (tid & 1) * 32;  // A: 32 contiguous floats
  const int bcol = tid & 127, bkh = (tid >> 7) * 32; // B: 32 k's of one column
  const float* Aptr = A + (size_t)(rowBase + arow) * K + acol;

  for (int kb = 0; kb < K; kb += BK) {
    if (kb + BK < K) {   // WGP-scope prefetch of next K-panel
      __builtin_prefetch(Aptr + kb + BK, 0, 3);
      __builtin_prefetch(W + (size_t)(kb + BK + bkh) * N + colBase + bcol, 0, 3);
    }
    // stage A: 8x float4 loads -> 4x b128 LDS stores
    {
      __align__(16) bf16 t[32];
#pragma unroll
      for (int i = 0; i < 8; ++i) {
        float4 f = *(const float4*)(Aptr + kb + i * 4);
        t[i * 4 + 0] = f2bf(f.x); t[i * 4 + 1] = f2bf(f.y);
        t[i * 4 + 2] = f2bf(f.z); t[i * 4 + 3] = f2bf(f.w);
      }
#pragma unroll
      for (int i = 0; i < 4; ++i)
        *(v8bf*)&As[arow][acol + i * 8] = *(const v8bf*)&t[i * 8];
    }
    // stage B transposed: 32 coalesced b32 loads -> 4x b128 LDS stores
    {
      __align__(16) bf16 t[32];
#pragma unroll
      for (int i = 0; i < 32; ++i)
        t[i] = f2bf(W[(size_t)(kb + bkh + i) * N + colBase + bcol]);
#pragma unroll
      for (int i = 0; i < 4; ++i)
        *(v8bf*)&BsT[bcol][bkh + i * 8] = *(const v8bf*)&t[i * 8];
    }
    __syncthreads();

#pragma unroll
    for (int ks = 0; ks < 2; ++ks) {
      v16bf afr[2], bfr[4];
#pragma unroll
      for (int mt = 0; mt < 2; ++mt)
        afr[mt] = frag_ld(&As[wm * 32 + mt * 16 + lrow][ks * 32 + khalf]);
#pragma unroll
      for (int nt = 0; nt < 4; ++nt)
        bfr[nt] = frag_ld(&BsT[wn * 64 + nt * 16 + lrow][ks * 32 + khalf]);
#pragma unroll
      for (int mt = 0; mt < 2; ++mt)
#pragma unroll
        for (int nt = 0; nt < 4; ++nt)
          acc[mt][nt] = __builtin_amdgcn_wmma_f32_16x16x32_bf16(
              false, afr[mt], false, bfr[nt], (short)0, acc[mt][nt], false, false);
    }
    __syncthreads();
  }

  // Epilogue. C/D layout: VGPR jj, lane l -> row = jj + 8*(l>>4), col = l&15.
#pragma unroll
  for (int mt = 0; mt < 2; ++mt) {
#pragma unroll
    for (int nt = 0; nt < 4; ++nt) {
      const int col = colBase + wn * 64 + nt * 16 + (lane & 15);
      const float bv = bias[col];
#pragma unroll
      for (int jj = 0; jj < 8; ++jj) {
        const int row = rowBase + wm * 32 + mt * 16 + (lane >> 4) * 8 + jj;
        float v = acc[mt][nt][jj] + bv;
        if (ACT == 1)      v = (v > 0.f) ? v + 1.f : __expf(v);       // elu(x)+1
        else if (ACT == 2) v = 0.5f * v * (1.f + erff(v * 0.70710678118654752440f));
        if (RES) v += res[(size_t)row * N + col];
        C[(size_t)row * N + col] = v;
      }
    }
  }
}

// ---------------------------------------------------------------------------
// Chunked causal linear attention, chunk = 64 tokens, one block per (b,h).
// KV state (64x64 f32) lives in per-wave WMMA accumulator registers.
// Per chunk:  S = Qf Kf^T (masked)     den = rowsum(S) + Qf.z + eps
//             O = S V + Qf KV ; out = O/den ; KV += Kf^T V ; z += colsum(Kf)
// ---------------------------------------------------------------------------
#define CH 64
#define CPITCH (CH + 8)   // 72 bf16 = 144B rows (16B aligned)

__global__ __launch_bounds__(128)
void linattn_chunk_wmma(const float* __restrict__ Q, const float* __restrict__ K,
                        const float* __restrict__ V, float* __restrict__ O,
                        int Bn, int Sn, int Dn, int Hh, int Dh) {
  __shared__ bf16 Qs[CH][CPITCH];    // [token][d]
  __shared__ bf16 Ks[CH][CPITCH];    // [token][d]
  __shared__ bf16 KsT[CH][CPITCH];   // [d][token]
  __shared__ bf16 VsT[CH][CPITCH];   // [d][token]
  __shared__ bf16 Ss[CH][CPITCH];    // [i][token] (masked S)
  __shared__ bf16 KVbT[CH][CPITCH];  // [n][d] bf16 snapshot of KV^T
  __shared__ float zs[CH];
  __shared__ float den[CH];

  const int tid   = threadIdx.x;
  const int lane  = tid & 31;
  const int w     = tid >> 5;          // 4 waves, each owns 16 output rows
  const int lrow  = lane & 15;
  const int khalf = (lane >> 4) * 8;
  const int bh = blockIdx.x;
  const int b = bh / Hh, hh = bh % Hh;
  const size_t base = (size_t)b * Sn * Dn + (size_t)hh * Dh;

  v8f kvacc[4];                        // KV state: wave w = d-rows [16w,16w+16)
#pragma unroll
  for (int nt = 0; nt < 4; ++nt) kvacc[nt] = v8f_zero();
  if (tid < CH) zs[tid] = 0.f;
  __syncthreads();

  for (int c = 0; c < Sn / CH; ++c) {
    const int t0 = c * CH;
    // ---- stage Q,K token-major (row r = t>>1, 32 contiguous dims) ----
    {
      const int r = tid >> 1, c0 = (tid & 1) * 32;
      __align__(16) bf16 tq[32], tk[32];
#pragma unroll
      for (int i = 0; i < 8; ++i) {
        const size_t off = base + (size_t)(t0 + r) * Dn + c0 + i * 4;
        float4 q4 = *(const float4*)(Q + off);
        float4 k4 = *(const float4*)(K + off);
        tq[i * 4 + 0] = f2bf(q4.x); tq[i * 4 + 1] = f2bf(q4.y);
        tq[i * 4 + 2] = f2bf(q4.z); tq[i * 4 + 3] = f2bf(q4.w);
        tk[i * 4 + 0] = f2bf(k4.x); tk[i * 4 + 1] = f2bf(k4.y);
        tk[i * 4 + 2] = f2bf(k4.z); tk[i * 4 + 3] = f2bf(k4.w);
      }
#pragma unroll
      for (int i = 0; i < 4; ++i) {
        *(v8bf*)&Qs[r][c0 + i * 8] = *(const v8bf*)&tq[i * 8];
        *(v8bf*)&Ks[r][c0 + i * 8] = *(const v8bf*)&tk[i * 8];
      }
    }
    // ---- stage K,V d-major (row d = tid&63, 32 contiguous tokens) ----
    {
      const int d = tid & 63, tt0 = (tid >> 6) * 32;
      __align__(16) bf16 tk[32], tv[32];
#pragma unroll
      for (int i = 0; i < 32; ++i) {
        const size_t off = base + (size_t)(t0 + tt0 + i) * Dn + d;
        tk[i] = f2bf(K[off]);
        tv[i] = f2bf(V[off]);
      }
#pragma unroll
      for (int i = 0; i < 4; ++i) {
        *(v8bf*)&KsT[d][tt0 + i * 8] = *(const v8bf*)&tk[i * 8];
        *(v8bf*)&VsT[d][tt0 + i * 8] = *(const v8bf*)&tv[i * 8];
      }
    }
    // ---- snapshot KV state (registers) -> KVbT[n][d] bf16 ----
#pragma unroll
    for (int nt = 0; nt < 4; ++nt)
#pragma unroll
      for (int jj = 0; jj < 8; ++jj) {
        const int m = w * 16 + (lane >> 4) * 8 + jj;   // d index
        const int n = nt * 16 + (lane & 15);
        KVbT[n][m] = f2bf(kvacc[nt][jj]);
      }
    __syncthreads();

    // ---- S = Qf @ Kf^T over d, causal mask on write ----
#pragma unroll
    for (int nt = 0; nt < 4; ++nt) {
      v8f sacc = v8f_zero();
#pragma unroll
      for (int ks = 0; ks < 2; ++ks) {
        v16bf a = frag_ld(&Qs[w * 16 + lrow][ks * 32 + khalf]);
        v16bf bb = frag_ld(&Ks[nt * 16 + lrow][ks * 32 + khalf]);
        sacc = __builtin_amdgcn_wmma_f32_16x16x32_bf16(false, a, false, bb,
                                                       (short)0, sacc, false, false);
      }
#pragma unroll
      for (int jj = 0; jj < 8; ++jj) {
        const int i = w * 16 + (lane >> 4) * 8 + jj;
        const int j = nt * 16 + (lane & 15);
        Ss[i][j] = f2bf((j <= i) ? sacc[jj] : 0.f);
      }
    }
    __syncthreads();

    // ---- denominator: rowsum(masked S) + Qf.z + eps ----
    if (tid < CH) {
      float rs = 0.f, qz = 0.f;
      for (int j = 0; j < CH; ++j) rs += bf2f(Ss[tid][j]);
      for (int d = 0; d < CH; ++d) qz += bf2f(Qs[tid][d]) * zs[d];
      den[tid] = rs + qz + ATT_EPS;
    }

    // ---- O = S @ V (over tokens) + Qf @ KV (over d) ----
    v8f oacc[4];
#pragma unroll
    for (int nt = 0; nt < 4; ++nt) oacc[nt] = v8f_zero();
#pragma unroll
    for (int nt = 0; nt < 4; ++nt) {
#pragma unroll
      for (int ks = 0; ks < 2; ++ks) {
        v16bf a = frag_ld(&Ss[w * 16 + lrow][ks * 32 + khalf]);
        v16bf bb = frag_ld(&VsT[nt * 16 + lrow][ks * 32 + khalf]);
        oacc[nt] = __builtin_amdgcn_wmma_f32_16x16x32_bf16(false, a, false, bb,
                                                           (short)0, oacc[nt], false, false);
        a  = frag_ld(&Qs[w * 16 + lrow][ks * 32 + khalf]);
        bb = frag_ld(&KVbT[nt * 16 + lrow][ks * 32 + khalf]);
        oacc[nt] = __builtin_amdgcn_wmma_f32_16x16x32_bf16(false, a, false, bb,
                                                           (short)0, oacc[nt], false, false);
      }
    }
    __syncthreads();   // den ready; Ss/KVbT consumed

    // ---- write out = O / den ----
#pragma unroll
    for (int nt = 0; nt < 4; ++nt)
#pragma unroll
      for (int jj = 0; jj < 8; ++jj) {
        const int i = w * 16 + (lane >> 4) * 8 + jj;
        const int d = nt * 16 + (lane & 15);
        O[base + (size_t)(t0 + i) * Dn + d] = oacc[nt][jj] / den[i];
      }

    // ---- KV += Kf^T @ V (contraction over tokens), stays in registers ----
#pragma unroll
    for (int nt = 0; nt < 4; ++nt) {
#pragma unroll
      for (int ks = 0; ks < 2; ++ks) {
        v16bf a = frag_ld(&KsT[w * 16 + lrow][ks * 32 + khalf]);
        v16bf bb = frag_ld(&VsT[nt * 16 + lrow][ks * 32 + khalf]);
        kvacc[nt] = __builtin_amdgcn_wmma_f32_16x16x32_bf16(false, a, false, bb,
                                                            (short)0, kvacc[nt], false, false);
      }
    }
    // ---- z += colsum(Kf) (contiguous in KsT) ----
    if (tid < CH) {
      float s = 0.f;
      for (int j = 0; j < CH; ++j) s += bf2f(KsT[tid][j]);
      zs[tid] += s;
    }
    __syncthreads();
  }
}

// ---------------------------------------------------------------------------
// LayerNorm over D=512, one wave32 per row (16 elems/lane, shfl_xor reduce).
// ---------------------------------------------------------------------------
__global__ __launch_bounds__(256)
void layernorm_rows(const float* __restrict__ X, const float* __restrict__ g,
                    const float* __restrict__ be, float* __restrict__ Y) {
  const int lane = threadIdx.x & 31;
  const int w = threadIdx.x >> 5;
  const int row = blockIdx.x * 8 + w;
  const float* x = X + (size_t)row * 512;
  float e[16], s = 0.f, ss = 0.f;
#pragma unroll
  for (int i = 0; i < 16; ++i) {
    e[i] = x[i * 32 + lane];
    s += e[i]; ss += e[i] * e[i];
  }
#pragma unroll
  for (int off = 16; off > 0; off >>= 1) {
    s  += __shfl_xor(s, off, 32);
    ss += __shfl_xor(ss, off, 32);
  }
  const float mean = s * (1.f / 512.f);
  const float var  = ss * (1.f / 512.f) - mean * mean;
  const float rstd = rsqrtf(var + LN_EPS);
  float* y = Y + (size_t)row * 512;
#pragma unroll
  for (int i = 0; i < 16; ++i) {
    const int col = i * 32 + lane;
    y[col] = (e[i] - mean) * rstd * g[col] + be[col];
  }
}

// ---------------------------------------------------------------------------
// Host orchestration (all on `stream`, graph-capture safe).
// ---------------------------------------------------------------------------
extern "C" void kernel_launch(void* const* d_in, const int* in_sizes, int n_in,
                              void* d_out, int out_size, void* d_ws, size_t ws_size,
                              hipStream_t stream) {
  constexpr int Lc = 4, Dc = 512, Hc = 8, DHc = 64, DFc = 2048, Bc = 4, Sc = 2048;
  constexpr int Mc = Bc * Sc;                   // 8192 tokens
  const size_t MD = (size_t)Mc * Dc;

  const float* h  = (const float*)d_in[0];
  const float* Wq = (const float*)d_in[1];  const float* bq = (const float*)d_in[2];
  const float* Wk = (const float*)d_in[3];  const float* bk = (const float*)d_in[4];
  const float* Wv = (const float*)d_in[5];  const float* bv = (const float*)d_in[6];
  const float* Wo = (const float*)d_in[7];  const float* bo = (const float*)d_in[8];
  const float* W1 = (const float*)d_in[9];  const float* b1 = (const float*)d_in[10];
  const float* W2 = (const float*)d_in[11]; const float* b2 = (const float*)d_in[12];
  const float* g1 = (const float*)d_in[13]; const float* be1 = (const float*)d_in[14];
  const float* g2 = (const float*)d_in[15]; const float* be2 = (const float*)d_in[16];
  const float* gf = (const float*)d_in[17]; const float* bef = (const float*)d_in[18];
  float* out = (float*)d_out;

  // Workspace: X | P0 P1 P2 P3. P0..P3 (4*M*D == M*DF) doubles as FFN hidden.
  float* X  = (float*)d_ws;
  float* Qb = X + MD;
  float* Kb = Qb + MD;
  float* Vb = Kb + MD;
  float* Ab = Vb + MD;
  float* Hb = Qb;

  hipMemcpyAsync(X, h, MD * sizeof(float), hipMemcpyDeviceToDevice, stream);

  const dim3 blk(256);
  const dim3 gD(Dc / BN, Mc / BM);
  const dim3 gF(DFc / BN, Mc / BM);
  const dim3 gLN(Mc / 8);

  for (int l = 0; l < Lc; ++l) {
    const float* wq = Wq + (size_t)l * Dc * Dc;  const float* bql = bq + (size_t)l * Dc;
    const float* wk = Wk + (size_t)l * Dc * Dc;  const float* bkl = bk + (size_t)l * Dc;
    const float* wv = Wv + (size_t)l * Dc * Dc;  const float* bvl = bv + (size_t)l * Dc;
    const float* wo = Wo + (size_t)l * Dc * Dc;  const float* bol = bo + (size_t)l * Dc;
    const float* w1 = W1 + (size_t)l * Dc * DFc; const float* b1l = b1 + (size_t)l * DFc;
    const float* w2 = W2 + (size_t)l * DFc * Dc; const float* b2l = b2 + (size_t)l * Dc;

    gemm_bias_act_wmma<1, false><<<gD, blk, 0, stream>>>(X, wq, bql, nullptr, Qb, Mc, Dc, Dc);
    gemm_bias_act_wmma<1, false><<<gD, blk, 0, stream>>>(X, wk, bkl, nullptr, Kb, Mc, Dc, Dc);
    gemm_bias_act_wmma<0, false><<<gD, blk, 0, stream>>>(X, wv, bvl, nullptr, Vb, Mc, Dc, Dc);
    linattn_chunk_wmma<<<dim3(Bc * Hc), dim3(128), 0, stream>>>(Qb, Kb, Vb, Ab,
                                                                Bc, Sc, Dc, Hc, DHc);
    gemm_bias_act_wmma<0, true><<<gD, blk, 0, stream>>>(Ab, wo, bol, X, X, Mc, Dc, Dc);
    layernorm_rows<<<gLN, blk, 0, stream>>>(X, g1 + (size_t)l * Dc, be1 + (size_t)l * Dc, X);
    gemm_bias_act_wmma<2, false><<<gF, blk, 0, stream>>>(X, w1, b1l, nullptr, Hb, Mc, DFc, Dc);
    gemm_bias_act_wmma<0, true><<<gD, blk, 0, stream>>>(Hb, w2, b2l, X, X, Mc, Dc, DFc);
    layernorm_rows<<<gLN, blk, 0, stream>>>(X, g2 + (size_t)l * Dc, be2 + (size_t)l * Dc, X);
  }
  layernorm_rows<<<gLN, blk, 0, stream>>>(X, gf, bef, out);
}
